// moegpt_71605694759040
// MI455X (gfx1250) — compile-verified, matrix operand order checked by reading, not run on hardware
//
#include <hip/hip_runtime.h>
#include <hip/hip_bf16.h>

typedef _Float16 half_t;
typedef __attribute__((ext_vector_type(16))) _Float16 v16h;
typedef __attribute__((ext_vector_type(8)))  _Float16 v8h;
typedef __attribute__((ext_vector_type(8)))  float    v8f;

#define NTOK 2048
#define HID  768
#define LRD  384
#define NH   12
#define FF   3072
#define NE   8
#define NV   50257
#define NL   2
#define WIN  1024
#define EOSD 50256

__device__ __forceinline__ int kpat(int vv, int hh) {
  return 2 * vv + (vv >= 4 ? 8 : 0) + hh * 8;
}

// chunk permutation for fragment-major LDS layout within a 32-wide K block:
// 8-chunks [0,1,2,3] stored at positions [0,2,1,3] so each lane-half's 16
// fragment elements are contiguous (k = {0..7,16..23} | {8..15,24..31}).
__device__ __forceinline__ int permc(int j) { return (j == 1) ? 2 : (j == 2) ? 1 : j; }

// 32B fragment load with 16B alignment guarantee (two ds_load_b128)
__device__ __forceinline__ v16h ld16(const half_t* p) {
  struct __attribute__((packed, aligned(16))) W { v16h v; };
  return ((const W*)p)->v;
}

// ---------------------------------------------------------------------------
// WMMA GEMM: C[M,Nn] (+=|=) act(A[M,K] @ B[K,Nn] + bias)
// 256x128 block, 8 waves x (64x64), BK=32, double-buffered LDS,
// branch-free clamped staging (OOB rows/cols masked only at the C store).
// ---------------------------------------------------------------------------
__global__ __launch_bounds__(256)
void gemm_wmma(const float* __restrict__ A, const float* __restrict__ B,
               const float* __restrict__ bias, float* __restrict__ C,
               int M, int K, int Nn, int act, int addc,
               const int* __restrict__ rowidx, const int* __restrict__ cntp)
{
  __shared__ half_t As[2][256][40];
  __shared__ half_t BsT[2][128][40];

  const int Meff = cntp ? *cntp : M;
  const int row0 = blockIdx.y * 256;
  const int col0 = blockIdx.x * 128;
  if (row0 >= Meff) return;

  const int tid  = threadIdx.x;
  const int lane = tid & 31;
  const int wave = tid >> 5;
  const int hh = lane >> 4, lr = lane & 15;
  const int wm = wave >> 1, wn = wave & 1;   // 4x2 waves, 64x64 each

  v8f acc[4][4];
#pragma unroll
  for (int i = 0; i < 4; i++)
#pragma unroll
    for (int j = 0; j < 4; j++)
#pragma unroll
      for (int r = 0; r < 8; r++) acc[i][j][r] = 0.f;

  // clamped A source row: rows >= Meff re-read a valid row; masked at store
  int arow = row0 + tid; if (arow >= Meff) arow = Meff - 1;
  const long asrc = rowidx ? (long)rowidx[arow] : (long)arow;
  const float* Abase = A + asrc * (long)K;

  // clamped B column
  const int bk0 = (tid & 1) * 16;
  int gcol = col0 + (tid >> 1); if (gcol >= Nn) gcol = Nn - 1;
  const float* Bbase = B + gcol;
  const int bcol = tid >> 1;

  float4 fa[8];
  float  fb[16];

  auto loadT = [&](int k0) {
#pragma unroll
    for (int c = 0; c < 8; c++) fa[c] = *(const float4*)(Abase + k0 + c * 4);
#pragma unroll
    for (int kk = 0; kk < 16; kk++) fb[kk] = Bbase[(long)(k0 + bk0 + kk) * Nn];
    __builtin_prefetch(Abase + k0 + 64, 0, 1);   // global_prefetch next-next tile
  };
  auto storeT = [&](int bi) {
#pragma unroll
    for (int j = 0; j < 4; j++) {
      v8h v;
#pragma unroll
      for (int i = 0; i < 8; i++) v[i] = (half_t)(((const float*)fa)[j * 8 + i]);
      *(v8h*)&As[bi][tid][permc(j) * 8] = v;
    }
#pragma unroll
    for (int j = 0; j < 2; j++) {
      v8h v;
#pragma unroll
      for (int i = 0; i < 8; i++) v[i] = (half_t)fb[j * 8 + i];
      *(v8h*)&BsT[bi][bcol][permc((bk0 >> 3) + j) * 8] = v;
    }
  };

  loadT(0);
  storeT(0);
  __syncthreads();

  const int nk = K >> 5;
  for (int t = 0; t < nk; t++) {
    const bool more = (t + 1) < nk;
    if (more) loadT((t + 1) * 32);       // global loads in flight over compute

    const int bi = t & 1;
    v16h af[4], bf[4];
#pragma unroll
    for (int mf = 0; mf < 4; mf++)
      af[mf] = ld16(&As[bi][wm * 64 + mf * 16 + lr][16 * hh]);
#pragma unroll
    for (int nf = 0; nf < 4; nf++)
      bf[nf] = ld16(&BsT[bi][wn * 64 + nf * 16 + lr][16 * hh]);
#pragma unroll
    for (int mf = 0; mf < 4; mf++)
#pragma unroll
      for (int nf = 0; nf < 4; nf++)
        acc[mf][nf] = __builtin_amdgcn_wmma_f32_16x16x32_f16(
            false, af[mf], false, bf[nf], (short)0, acc[mf][nf], false, false);

    if (more) storeT(bi ^ 1);
    __syncthreads();
  }

#pragma unroll
  for (int mf = 0; mf < 4; mf++)
#pragma unroll
    for (int nf = 0; nf < 4; nf++)
#pragma unroll
      for (int r = 0; r < 8; r++) {
        int row = row0 + wm * 64 + mf * 16 + r + 8 * hh;
        int col = col0 + wn * 64 + nf * 16 + lr;
        if (row < Meff && col < Nn) {
          float v = acc[mf][nf][r];
          if (bias) v += bias[col];
          if (act == 1) { float tt = v > 0.f ? v : 0.f; v = tt * tt; }
          long idx = (long)row * Nn + col;
          if (addc) v += C[idx];
          C[idx] = v;
        }
      }
}

// ---------------------------------------------------------------------------
// Flash attention: one head x 64 query rows per block (4 waves x 16 rows).
// WMMA for S=QK^T and O+=P.V, online softmax, causal+window+doc mask.
// ---------------------------------------------------------------------------
__global__ __launch_bounds__(128)
void attn_wmma(const float* __restrict__ qc, const float* __restrict__ qrr,
               const float* __restrict__ kc, const float* __restrict__ krr,
               const float* __restrict__ vb, const int* __restrict__ docs,
               float* __restrict__ o)
{
  __shared__ half_t Ks[64][136];    // [key][d], fragment-major within 32-d blocks
  __shared__ half_t VsT[128][72];   // [d][key], fragment-major within 32-key blocks
  __shared__ half_t Ps[4][16][72];  // per-wave P scratch, fragment-major

  const int h  = blockIdx.y;
  const int qb = blockIdx.x;
  const int tid = threadIdx.x;
  const int wave = tid >> 5, lane = tid & 31;
  const int hh = lane >> 4, lr = lane & 15;
  const int qrow0 = qb * 64 + wave * 16;
  const float scale = 0.08838834764831845f; // 1/sqrt(128)

  v16h qf[4];
  {
    const long qn = qrow0 + lr;
#pragma unroll
    for (int kf = 0; kf < 4; kf++)
#pragma unroll
      for (int vv = 0; vv < 8; vv++) {
        int d = kf * 32 + kpat(vv, hh);
        float a0 = (d     < 64) ? qc[qn * HID + h * 64 + d]     : qrr[qn * 64 + d - 64];
        float a1 = (d + 1 < 64) ? qc[qn * HID + h * 64 + d + 1] : qrr[qn * 64 + d + 1 - 64];
        qf[kf][2 * vv] = (half_t)a0; qf[kf][2 * vv + 1] = (half_t)a1;
      }
  }
  int mydoc[8];
#pragma unroll
  for (int r = 0; r < 8; r++) mydoc[r] = docs[qrow0 + r + 8 * hh];

  float m[8], l[8];
  v8f oac[8];
#pragma unroll
  for (int r = 0; r < 8; r++) { m[r] = -1e30f; l[r] = 0.f; }
#pragma unroll
  for (int of = 0; of < 8; of++)
#pragma unroll
    for (int r = 0; r < 8; r++) oac[of][r] = 0.f;

  int kt0 = qb * 64 - (WIN - 1); kt0 = (kt0 < 0) ? 0 : (kt0 >> 6);
  for (int kt = kt0; kt <= qb; kt++) {
    { // K tile: thread owns (key row, 64-d half); permuted v8h stores
      int row = tid >> 1, c0 = (tid & 1) * 64;
      long kn = kt * 64 + row;
#pragma unroll
      for (int cc = 0; cc < 8; cc++) {
        int d8 = c0 + cc * 8;
        v8h v;
#pragma unroll
        for (int i = 0; i < 8; i++) {
          int d = d8 + i;
          v[i] = (half_t)((d < 64) ? kc[kn * HID + h * 64 + d] : krr[kn * 64 + d - 64]);
        }
        *(v8h*)&Ks[row][(d8 >> 5) * 32 + permc((d8 >> 3) & 3) * 8] = v;
      }
    }
    { // V tile transposed: thread owns d column, 64 keys; permuted v8h stores
      int d = tid;
#pragma unroll
      for (int cc = 0; cc < 8; cc++) {
        int k8 = cc * 8;
        v8h v;
#pragma unroll
        for (int i = 0; i < 8; i++)
          v[i] = (half_t)vb[(long)(kt * 64 + k8 + i) * 1536 + h * 128 + d];
        *(v8h*)&VsT[d][(k8 >> 5) * 32 + permc((k8 >> 3) & 3) * 8] = v;
      }
    }
    __syncthreads();

    // S = Q K^T
    v8f sac[4];
#pragma unroll
    for (int nf = 0; nf < 4; nf++) {
#pragma unroll
      for (int r = 0; r < 8; r++) sac[nf][r] = 0.f;
      int col = nf * 16 + lr;
#pragma unroll
      for (int kf = 0; kf < 4; kf++) {
        v16h bf = ld16(&Ks[col][kf * 32 + 16 * hh]);
        sac[nf] = __builtin_amdgcn_wmma_f32_16x16x32_f16(
            false, qf[kf], false, bf, (short)0, sac[nf], false, false);
      }
    }

    float sv[4][8];
#pragma unroll
    for (int nf = 0; nf < 4; nf++) {
      int key = kt * 64 + nf * 16 + lr;
      int kd = docs[key];
#pragma unroll
      for (int r = 0; r < 8; r++) {
        int qi = qrow0 + r + 8 * hh;
        bool ok = (key <= qi) && (qi - key < WIN) && (kd == mydoc[r]);
        sv[nf][r] = ok ? sac[nf][r] * scale : -1e30f;
      }
    }

    float mn[8], al[8], rs[8];
#pragma unroll
    for (int r = 0; r < 8; r++) {
      float x = -1e30f;
#pragma unroll
      for (int nf = 0; nf < 4; nf++) x = fmaxf(x, sv[nf][r]);
#pragma unroll
      for (int off = 1; off < 16; off <<= 1) x = fmaxf(x, __shfl_xor(x, off, 16));
      mn[r] = fmaxf(m[r], x);
      al[r] = __expf(m[r] - mn[r]);
      rs[r] = 0.f;
    }
#pragma unroll
    for (int nf = 0; nf < 4; nf++) {
      int a = nf & 1;
#pragma unroll
      for (int r = 0; r < 8; r++) {
        float s = sv[nf][r];
        float p = (s <= -1e29f) ? 0.f : __expf(s - mn[r]);
        rs[r] += p;
        int cc = permc(2 * a + (lr >> 3));
        Ps[wave][r + 8 * hh][(nf >> 1) * 32 + cc * 8 + (lr & 7)] = (half_t)p;
      }
    }
#pragma unroll
    for (int r = 0; r < 8; r++) {
      float x = rs[r];
#pragma unroll
      for (int off = 1; off < 16; off <<= 1) x += __shfl_xor(x, off, 16);
      l[r] = l[r] * al[r] + x;
      m[r] = mn[r];
    }
#pragma unroll
    for (int of = 0; of < 8; of++)
#pragma unroll
      for (int r = 0; r < 8; r++) oac[of][r] *= al[r];

    v16h pa[2];
#pragma unroll
    for (int kf = 0; kf < 2; kf++)
      pa[kf] = ld16(&Ps[wave][lr][kf * 32 + 16 * hh]);
#pragma unroll
    for (int of = 0; of < 8; of++) {
      int d = of * 16 + lr;
#pragma unroll
      for (int kf = 0; kf < 2; kf++) {
        v16h bf = ld16(&VsT[d][kf * 32 + 16 * hh]);
        oac[of] = __builtin_amdgcn_wmma_f32_16x16x32_f16(
            false, pa[kf], false, bf, (short)0, oac[of], false, false);
      }
    }
    __syncthreads();
  }

#pragma unroll
  for (int r = 0; r < 8; r++) {
    float inv = (l[r] > 0.f) ? 1.f / l[r] : 0.f;
    long n = qrow0 + r + 8 * hh;
#pragma unroll
    for (int of = 0; of < 8; of++) {
      int d = of * 16 + lr;
      o[n * 1536 + h * 128 + d] = oac[of][r] * inv;
    }
  }
}

// ------------------------- small elementwise kernels -----------------------
__global__ void docs_kernel(const int* __restrict__ tok, int* __restrict__ docs) {
  int i = blockIdx.x * blockDim.x + threadIdx.x;
  if (i >= NTOK) return;
  int c = 0;
  for (int j = 0; j <= i; j++) c += (tok[j] == EOSD);
  docs[i] = c;
}

__global__ void embed_rms_kernel(const int* __restrict__ tok,
                                 const float* __restrict__ embd, float* __restrict__ x) {
  int n = blockIdx.x * blockDim.x + threadIdx.x;
  if (n >= NTOK) return;
  const float* e = embd + (long)tok[n] * HID;
  float s = 0.f;
  for (int j = 0; j < HID; j++) { float v = e[j]; s += v * v; }
  float sc = rsqrtf(s / (float)HID + 1.1920929e-7f);
  float* xo = x + (long)n * HID;
  for (int j = 0; j < HID; j++) xo[j] = e[j] * sc;
}

__global__ void rms_kernel(const float* __restrict__ x, float* __restrict__ y) {
  int n = blockIdx.x * blockDim.x + threadIdx.x;
  if (n >= NTOK) return;
  const float* xi = x + (long)n * HID;
  float s = 0.f;
  for (int j = 0; j < HID; j++) { float v = xi[j]; s += v * v; }
  float sc = rsqrtf(s / (float)HID + 1.1920929e-7f);
  float* yo = y + (long)n * HID;
  for (int j = 0; j < HID; j++) yo[j] = xi[j] * sc;
}

__global__ void ln_kernel(const float* __restrict__ x, const float* __restrict__ w,
                          const float* __restrict__ b, float* __restrict__ y) {
  int n = blockIdx.x * blockDim.x + threadIdx.x;
  if (n >= NTOK) return;
  const float* xi = x + (long)n * HID;
  float s1 = 0.f, s2 = 0.f;
  for (int j = 0; j < HID; j++) { float v = xi[j]; s1 += v; s2 += v * v; }
  float mu = s1 / (float)HID;
  float var = s2 / (float)HID - mu * mu;
  float sc = rsqrtf(var + 1e-5f);
  float* yo = y + (long)n * HID;
  for (int j = 0; j < HID; j++) yo[j] = (xi[j] - mu) * sc * w[j] + b[j];
}

__global__ void rope_kernel(const float* __restrict__ src, float* __restrict__ dst) {
  int n = blockIdx.x * blockDim.x + threadIdx.x;
  if (n >= NTOK) return;
  const float* s = src + (long)n * 64;
  float* d = dst + (long)n * 64;
  for (int j = 0; j < 32; j++) {
    float inv = expf(-(float)j * (9.210340371976184f / 32.f));
    float a = (float)n * inv;
    float c = cosf(a), si = sinf(a);
    float x0 = s[j], x1 = s[j + 32];
    d[j]      = x0 * c - x1 * si;
    d[j + 32] = x1 * c + x0 * si;
  }
}

__global__ void zero_counts_kernel(int* __restrict__ counts) {
  if (threadIdx.x < NE) counts[threadIdx.x] = 0;
}

__global__ void router_kernel(const float* __restrict__ xn, const float* __restrict__ rw,
                              const float* __restrict__ rb, float* __restrict__ probs,
                              int* __restrict__ counts, int* __restrict__ perm,
                              float* __restrict__ wl) {
  int n = blockIdx.x * blockDim.x + threadIdx.x;
  if (n >= NTOK) return;
  float lg[NE];
#pragma unroll
  for (int e = 0; e < NE; e++) lg[e] = rb[e];
  const float* xr = xn + (long)n * HID;
  for (int j = 0; j < HID; j++) {
    float xv = xr[j];
    const float* w = rw + (long)j * NE;
#pragma unroll
    for (int e = 0; e < NE; e++) lg[e] += xv * w[e];
  }
  float mx = lg[0];
#pragma unroll
  for (int e = 1; e < NE; e++) mx = fmaxf(mx, lg[e]);
  float p[NE]; float sum = 0.f;
#pragma unroll
  for (int e = 0; e < NE; e++) { p[e] = __expf(lg[e] - mx); sum += p[e]; }
#pragma unroll
  for (int e = 0; e < NE; e++) { p[e] /= sum; probs[(long)n * NE + e] = p[e]; }
  int e0 = 0;
#pragma unroll
  for (int e = 1; e < NE; e++) if (p[e] > p[e0]) e0 = e;
  int e1 = (e0 == 0) ? 1 : 0;
#pragma unroll
  for (int e = 0; e < NE; e++) if (e != e0 && p[e] > p[e1]) e1 = e;
  float ws = p[e0] + p[e1];
  float w0 = p[e0] / ws, w1 = p[e1] / ws;
  int pos0 = atomicAdd(&counts[e0], 1);
  perm[e0 * NTOK + pos0] = n; wl[e0 * NTOK + pos0] = w0;
  int pos1 = atomicAdd(&counts[e1], 1);
  perm[e1 * NTOK + pos1] = n; wl[e1 * NTOK + pos1] = w1;
}

__global__ void balance_kernel(const float* __restrict__ probs, float* __restrict__ out_bl,
                               int init) {
  __shared__ float avg[NE];
  int t = threadIdx.x;
  if (t < NE) {
    float s = 0.f;
    for (int n = 0; n < NTOK; n++) s += probs[(long)n * NE + t];
    avg[t] = s / (float)NTOK;
  }
  __syncthreads();
  if (t == 0) {
    int cnt[NE];
    for (int e = 0; e < NE; e++) cnt[e] = 0;
    for (int n = 0; n < NTOK; n++) {
      int a = 0; float bv = probs[(long)n * NE];
      for (int e = 1; e < NE; e++) {
        float v = probs[(long)n * NE + e];
        if (v > bv) { bv = v; a = e; }
      }
      cnt[a]++;
    }
    float bl = 0.f;
    for (int e = 0; e < NE; e++) bl += ((float)cnt[e] / (float)NTOK) * avg[e];
    bl *= 0.001f;
    if (init) *out_bl = bl; else *out_bl += bl;
  }
}

__global__ void scatter_kernel(const float* __restrict__ u, const int* __restrict__ perm,
                               const float* __restrict__ wl, const int* __restrict__ cnt,
                               float* __restrict__ x) {
  long idx = (long)blockIdx.x * blockDim.x + threadIdx.x;
  int i = (int)(idx / HID), j = (int)(idx % HID);
  if (i >= *cnt) return;
  x[(long)perm[i] * HID + j] += wl[i] * u[(long)i * HID + j];
}

// ---------------------------------------------------------------------------
static inline void launch_gemm(const float* A, const float* B, const float* bias,
                               float* C, int M, int K, int Nn, int act, int addc,
                               const int* rowidx, const int* cntp, hipStream_t s) {
  dim3 g((Nn + 127) / 128, (M + 255) / 256);
  gemm_wmma<<<g, 256, 0, s>>>(A, B, bias, C, M, K, Nn, act, addc, rowidx, cntp);
}

extern "C" void kernel_launch(void* const* d_in, const int* in_sizes, int n_in,
                              void* d_out, int out_size, void* d_ws, size_t ws_size,
                              hipStream_t stream) {
  const int*   tokens = (const int*)d_in[0];
  const float* embd   = (const float*)d_in[1];
  const float* qd_w  = (const float*)d_in[2];  const float* qd_b  = (const float*)d_in[3];
  const float* qu_w  = (const float*)d_in[4];  const float* qu_b  = (const float*)d_in[5];
  const float* qr_w  = (const float*)d_in[6];  const float* qr_b  = (const float*)d_in[7];
  const float* kvd_w = (const float*)d_in[8];  const float* kvd_b = (const float*)d_in[9];
  const float* kup_w = (const float*)d_in[10]; const float* kup_b = (const float*)d_in[11];
  const float* kr_w  = (const float*)d_in[12]; const float* kr_b  = (const float*)d_in[13];
  const float* vup_w = (const float*)d_in[14]; const float* vup_b = (const float*)d_in[15];
  const float* o_w   = (const float*)d_in[16]; const float* o_b   = (const float*)d_in[17];
  const float* n1_w  = (const float*)d_in[18]; const float* n1_b  = (const float*)d_in[19];
  const float* n2_w  = (const float*)d_in[20]; const float* n2_b  = (const float*)d_in[21];
  const float* rt_w  = (const float*)d_in[22]; const float* rt_b  = (const float*)d_in[23];
  const float* fc1_w = (const float*)d_in[24]; const float* fc1_b = (const float*)d_in[25];
  const float* fc2_w = (const float*)d_in[26]; const float* fc2_b = (const float*)d_in[27];
  const float* out_w = (const float*)d_in[28]; const float* out_b = (const float*)d_in[29];

  float* logits = (float*)d_out;
  float* out_bl = logits + (size_t)NTOK * NV;

  float* W = (float*)d_ws;
  size_t off = 0;
  float* x    = W + off; off += (size_t)NTOK * HID;
  float* xn   = W + off; off += (size_t)NTOK * HID;
  float* qd   = W + off; off += (size_t)NTOK * LRD;
  float* q    = W + off; off += (size_t)NTOK * HID;
  float* qrb  = W + off; off += (size_t)NTOK * 64;
  float* qrr  = W + off; off += (size_t)NTOK * 64;
  float* kv   = W + off; off += (size_t)NTOK * LRD;
  float* k    = W + off; off += (size_t)NTOK * HID;
  float* krb  = W + off; off += (size_t)NTOK * 64;
  float* krr  = W + off; off += (size_t)NTOK * 64;
  float* vbuf = W + off; off += (size_t)NTOK * 2 * HID;
  float* ocat = W + off; off += (size_t)NTOK * 2 * HID;
  float* tbuf = W + off; off += (size_t)NTOK * FF;
  float* ubuf = W + off; off += (size_t)NTOK * HID;
  float* probs= W + off; off += (size_t)NTOK * NE;
  float* wl   = W + off; off += (size_t)NE * NTOK;
  int* ip     = (int*)(W + off);
  int* docs   = ip;
  int* counts = ip + NTOK;
  int* perm   = ip + NTOK + NE;

  const dim3 tb(256);
  const dim3 tgrid((NTOK + 255) / 256);

  docs_kernel<<<tgrid, tb, 0, stream>>>(tokens, docs);
  embed_rms_kernel<<<tgrid, tb, 0, stream>>>(tokens, embd, x);

  for (int l = 0; l < NL; l++) {
    ln_kernel<<<tgrid, tb, 0, stream>>>(x, n1_w + l * HID, n1_b + l * HID, xn);
    launch_gemm(xn, qd_w + (size_t)l * HID * LRD, qd_b + l * LRD, qd,
                NTOK, HID, LRD, 0, 0, nullptr, nullptr, stream);
    launch_gemm(qd, qu_w + (size_t)l * LRD * HID, qu_b + l * HID, q,
                NTOK, LRD, HID, 0, 0, nullptr, nullptr, stream);
    launch_gemm(xn, qr_w + (size_t)l * HID * 64, qr_b + l * 64, qrb,
                NTOK, HID, 64, 0, 0, nullptr, nullptr, stream);
    rope_kernel<<<tgrid, tb, 0, stream>>>(qrb, qrr);
    launch_gemm(xn, kvd_w + (size_t)l * HID * LRD, kvd_b + l * LRD, kv,
                NTOK, HID, LRD, 0, 0, nullptr, nullptr, stream);
    launch_gemm(kv, kup_w + (size_t)l * LRD * HID, kup_b + l * HID, k,
                NTOK, LRD, HID, 0, 0, nullptr, nullptr, stream);
    launch_gemm(xn, kr_w + (size_t)l * HID * 64, kr_b + l * 64, krb,
                NTOK, HID, 64, 0, 0, nullptr, nullptr, stream);
    rope_kernel<<<tgrid, tb, 0, stream>>>(krb, krr);
    launch_gemm(kv, vup_w + (size_t)l * LRD * 2 * HID, vup_b + l * 2 * HID, vbuf,
                NTOK, LRD, 2 * HID, 0, 0, nullptr, nullptr, stream);

    attn_wmma<<<dim3(NTOK / 64, NH), 128, 0, stream>>>(q, qrr, k, krr, vbuf, docs, ocat);

    launch_gemm(ocat, o_w + (size_t)l * 2 * HID * HID, o_b + l * HID, x,
                NTOK, 2 * HID, HID, 0, 1, nullptr, nullptr, stream);

    ln_kernel<<<tgrid, tb, 0, stream>>>(x, n2_w + l * HID, n2_b + l * HID, xn);
    zero_counts_kernel<<<1, 32, 0, stream>>>(counts);
    router_kernel<<<tgrid, tb, 0, stream>>>(xn, rt_w + (size_t)l * HID * NE,
                                            rt_b + l * NE, probs, counts, perm, wl);
    balance_kernel<<<1, 32, 0, stream>>>(probs, out_bl, l == 0 ? 1 : 0);

    for (int e = 0; e < NE; e++) {
      const float* w1 = fc1_w + ((size_t)(l * NE + e)) * HID * FF;
      const float* b1 = fc1_b + ((size_t)(l * NE + e)) * FF;
      const float* w2 = fc2_w + ((size_t)(l * NE + e)) * FF * HID;
      const float* b2 = fc2_b + ((size_t)(l * NE + e)) * HID;
      launch_gemm(xn, w1, b1, tbuf, NTOK, HID, FF, 1, 0,
                  perm + e * NTOK, counts + e, stream);
      launch_gemm(tbuf, w2, b2, ubuf, NTOK, FF, HID, 0, 0,
                  nullptr, counts + e, stream);
      scatter_kernel<<<(NTOK * HID) / 256, 256, 0, stream>>>(
          ubuf, perm + e * NTOK, wl + e * NTOK, counts + e, x);
    }
  }

  rms_kernel<<<tgrid, tb, 0, stream>>>(x, xn);
  launch_gemm(xn, out_w, out_b, logits, NTOK, HID, NV, 0, 0, nullptr, nullptr, stream);

  (void)in_sizes; (void)n_in; (void)out_size; (void)ws_size;
}
